// Linear_GCNFace_39376260169853
// MI455X (gfx1250) — compile-verified, matrix operand order checked by reading, not run on hardware
//
#include <hip/hip_runtime.h>

#define HDIM 32
#define EPS_BN 1e-5f

typedef __attribute__((ext_vector_type(16))) _Float16 v16h;
typedef __attribute__((ext_vector_type(8)))  float    v8f;
typedef __attribute__((ext_vector_type(2)))  float    v2f;

#if defined(__HIP_DEVICE_COMPILE__) && __has_builtin(__builtin_amdgcn_wmma_f32_16x16x4_f32)
#define USE_WMMA_F32 1
#else
#define USE_WMMA_F32 0
#endif

// ---------------------------------------------------------------------------
// Kernel 1: h = x @ W1 + b1 (N x 32) + BN sum/sumsq.
// One wave = 32 nodes. Lane j owns channel j: broadcast x via v_readlane,
// coalesced stores of h, register accumulation, 1 LDS atomic/lane/wave.
// ---------------------------------------------------------------------------
__global__ void k_h_stats(const float* __restrict__ x,
                          const float* __restrict__ W1,
                          const float* __restrict__ b1,
                          float* __restrict__ h,
                          float* __restrict__ stats,   // [0,32) sums, [32,64) sumsq
                          int n) {
  __shared__ float ls[HDIM];
  __shared__ float lq[HDIM];
  int tid  = threadIdx.x;
  int lane = tid & 31;
  int wave = tid >> 5;
  if (tid < HDIM) { ls[tid] = 0.f; lq[tid] = 0.f; }
  __syncthreads();

  int base = (blockIdx.x * (blockDim.x >> 5) + wave) * 32;  // 32 nodes per wave
  if (base < n) {
    int node = base + lane;
    float x0 = 0.f, x1 = 0.f;
    if (node < n) { x0 = x[node * 2 + 0]; x1 = x[node * 2 + 1]; }
    int j = lane;
    float w0j = W1[j], w1j = W1[HDIM + j], bj = b1[j];
    float sum = 0.f, sq = 0.f;
#pragma unroll
    for (int t = 0; t < 32; ++t) {
      if (base + t < n) {                       // uniform per wave
        float a0 = __shfl(x0, t, 32);           // v_readlane (t is constant)
        float a1 = __shfl(x1, t, 32);
        float v  = fmaf(a0, w0j, fmaf(a1, w1j, bj));
        h[(size_t)(base + t) * HDIM + j] = v;   // coalesced across lanes
        sum += v;
        sq   = fmaf(v, v, sq);
      }
    }
    atomicAdd(&ls[j], sum);
    atomicAdd(&lq[j], sq);
  }
  __syncthreads();
  if (tid < HDIM) {
    atomicAdd(&stats[tid], ls[tid]);
    atomicAdd(&stats[HDIM + tid], lq[tid]);
  }
}

// ---------------------------------------------------------------------------
// Kernel 2: inv_norm[n] = rsqrt(||face[n]||^2 + 1e-12). Wave per row.
// ---------------------------------------------------------------------------
__global__ void k_face_inv_norm(const float* __restrict__ face,
                                float* __restrict__ invn,
                                int n, int f4) {
  int wave = threadIdx.x >> 5;
  int lane = threadIdx.x & 31;
  int row  = blockIdx.x * (blockDim.x >> 5) + wave;
  if (row >= n) return;
  const float4* fp = (const float4*)(face) + (size_t)row * f4;
  float acc = 0.f;
  for (int i = lane; i < f4; i += 32) {
    float4 v = fp[i];
    acc = fmaf(v.x, v.x, acc);
    acc = fmaf(v.y, v.y, acc);
    acc = fmaf(v.z, v.z, acc);
    acc = fmaf(v.w, v.w, acc);
  }
#pragma unroll
  for (int m = 16; m >= 1; m >>= 1) acc += __shfl_xor(acc, m, 32);
  if (lane == 0) invn[row] = rsqrtf(acc + 1e-12f);
}

// ---------------------------------------------------------------------------
// Kernel 3 (1 wave): BN scale/shift, v = Wg @ wf, folded bias constants,
// f16 copy of W2 (used only by the f16 fallback WMMA path).
// stats: [0,64) sums|sumsq, [64,96) scale, [96,128) shift, [128,160) v,
//        [160]=cA, [161]=cS, [162]=cOut
// ---------------------------------------------------------------------------
__global__ void k_setup(const float* __restrict__ gamma, const float* __restrict__ beta,
                        const float* __restrict__ W2, const float* __restrict__ b2,
                        const float* __restrict__ w0, const float* __restrict__ b0,
                        const float* __restrict__ Wg, const float* __restrict__ bg,
                        const float* __restrict__ wf, const float* __restrict__ bf,
                        float* __restrict__ stats, _Float16* __restrict__ w2h,
                        float invN) {
  int j = threadIdx.x;  // 32 threads
  float mu  = stats[j] * invN;
  float var = fmaf(-mu, mu, stats[HDIM + j] * invN);   // biased var
  float sc  = rsqrtf(var + EPS_BN) * gamma[j];
  stats[64 + j] = sc;
  stats[96 + j] = fmaf(-mu, sc, beta[j]);
  float vj = 0.f;
  for (int k = 0; k < HDIM; ++k) vj = fmaf(Wg[j * HDIM + k], wf[k], vj);
  stats[128 + j] = vj;                                  // (Wg @ wf)[j]
  for (int k = 0; k < HDIM; ++k) w2h[j * HDIM + k] = (_Float16)W2[j * HDIM + k];
  __syncthreads();
  if (j == 0) {
    float cA = b0[0], cS = 0.f, cOut = bf[0];
    for (int k = 0; k < HDIM; ++k) {
      cA   = fmaf(b2[k], w0[k], cA);            // b0 + b2.w0
      cS   = fmaf(b2[k], stats[128 + k], cS);   // b2.v
      cOut = fmaf(bg[k], wf[k], cOut);          // bf + bg.wf
    }
    stats[160] = cA;
    stats[161] = cS;
    stats[162] = cOut;
  }
}

// ---------------------------------------------------------------------------
// Kernel 4: wave per edge: ew = cos(face[src], face[dst]); degree atomics.
// face (100MB) is L2-resident (192MB) after k_face_inv_norm warmed it.
// ---------------------------------------------------------------------------
__global__ void k_edge_dot(const int* __restrict__ ei,
                           const float* __restrict__ face,
                           const float* __restrict__ invn,
                           float* __restrict__ ew,
                           float* __restrict__ deg,
                           int e_total, int f4) {
  int wave = threadIdx.x >> 5;
  int lane = threadIdx.x & 31;
  int e = blockIdx.x * (blockDim.x >> 5) + wave;
  if (e >= e_total) return;
  int s = ei[e];
  int d = ei[e_total + e];
  const float4* fs = (const float4*)(face) + (size_t)s * f4;
  const float4* fd = (const float4*)(face) + (size_t)d * f4;
  float acc = 0.f;
  for (int i = lane; i < f4; i += 32) {
    float4 a = fs[i];
    float4 b = fd[i];
    acc = fmaf(a.x, b.x, acc);
    acc = fmaf(a.y, b.y, acc);
    acc = fmaf(a.z, b.z, acc);
    acc = fmaf(a.w, b.w, acc);
  }
#pragma unroll
  for (int m = 16; m >= 1; m >>= 1) acc += __shfl_xor(acc, m, 32);
  if (lane == 0) {
    float w = acc * invn[s] * invn[d];
    ew[e] = w;
    atomicAdd(&deg[d], w);          // consumed later as (deg + 1)
  }
}

// ---------------------------------------------------------------------------
// Kernel 5: WMMA. One wave = 16 rows: T = BN/PReLU(h) @ W2.
// Preferred path: exact f32 math via 8-step K=4 chain of
// v_wmma_f32_16x16x4_f32 per 16-column tile (16 WMMAs/wave).
// Fallback: single-K32 v_wmma_f32_16x16x32_f16 per tile.
// emb = T + b2 never hits memory; the accumulator is reduced across lanes to
// aPart = emb.w0 + b0 and s = emb.v (v = Wg@wf), constants pre-folded.
//
// f32 A 16x4: lane (m=L&15, half=L>>4) holds {A[m][kk+2h], A[m][kk+2h+1]}.
// B mirrors A with N in place of M. C/D: lane = col N=L&15, rows M=v+8*half.
// ---------------------------------------------------------------------------
__global__ void k_emb_wmma(const float* __restrict__ h,
                           const float* __restrict__ stats,
                           const float* __restrict__ W2,
                           const _Float16* __restrict__ w2h,
                           const float* __restrict__ w0,
                           const float* __restrict__ alpha,
                           float* __restrict__ aPart,
                           float* __restrict__ sArr,
                           int n) {
  int wave = threadIdx.x >> 5;
  int lane = threadIdx.x & 31;
  int row0 = (blockIdx.x * (blockDim.x >> 5) + wave) * 16;
  if (row0 >= n) return;   // uniform per wave: EXEC stays all-ones for WMMA

  const float* scale = stats + 64;
  const float* shift = stats + 96;
  const float* vvec  = stats + 128;
  float cA = stats[160];
  float cS = stats[161];
  float al = alpha[0];

  int m    = lane & 15;
  int half = lane >> 4;
  int row  = row0 + m;
  bool ok  = row < n;
  int ncol = lane & 15;

#if USE_WMMA_F32
  // ---- exact f32 path: 8 chained K=4 WMMAs per column tile ----
  const float2* hp2 = (const float2*)(h + (size_t)row * HDIM);
  const float2* sc2 = (const float2*)scale;
  const float2* sh2 = (const float2*)shift;
  float hh[16];
#pragma unroll
  for (int s = 0; s < 8; ++s) {
    int p = 2 * s + half;                 // float2 index: k = 2*p, 2*p+1
    float2 hv = ok ? hp2[p] : make_float2(0.f, 0.f);
    float2 sc = sc2[p];
    float2 sh = sh2[p];
    float a0 = fmaf(hv.x, sc.x, sh.x);    // BatchNorm
    float a1 = fmaf(hv.y, sc.y, sh.y);
    a0 = a0 >= 0.f ? a0 : al * a0;        // PReLU
    a1 = a1 >= 0.f ? a1 : al * a1;
    hh[2 * s + 0] = a0;
    hh[2 * s + 1] = a1;
  }
  float b0a[16], b1a[16];
#pragma unroll
  for (int s = 0; s < 8; ++s) {
    int k0 = 4 * s + 2 * half;
    b0a[2 * s + 0] = W2[(k0 + 0) * HDIM + ncol];
    b0a[2 * s + 1] = W2[(k0 + 1) * HDIM + ncol];
    b1a[2 * s + 0] = W2[(k0 + 0) * HDIM + ncol + 16];
    b1a[2 * s + 1] = W2[(k0 + 1) * HDIM + ncol + 16];
  }
  v8f t0 = {};
  v8f t1 = {};
#pragma unroll
  for (int s = 0; s < 8; ++s) {
    v2f a  = { hh[2 * s], hh[2 * s + 1] };
    v2f bA = { b0a[2 * s], b0a[2 * s + 1] };
    v2f bB = { b1a[2 * s], b1a[2 * s + 1] };
    t0 = __builtin_amdgcn_wmma_f32_16x16x4_f32(false, a, false, bA,
                                               (short)0, t0, false, false);
    t1 = __builtin_amdgcn_wmma_f32_16x16x4_f32(false, a, false, bB,
                                               (short)0, t1, false, false);
  }
#else
  // ---- f16 fallback: single K=32 WMMA per column tile ----
  int kb = half * 8;
  const float* hp = h + (size_t)row * HDIM;
  v16h A;
#pragma unroll
  for (int i = 0; i < 8; ++i) {
    int k0 = kb + i;
    int k1 = kb + 16 + i;
    float a0 = ok ? hp[k0] : 0.f;
    float a1 = ok ? hp[k1] : 0.f;
    a0 = fmaf(a0, scale[k0], shift[k0]);
    a1 = fmaf(a1, scale[k1], shift[k1]);
    a0 = a0 >= 0.f ? a0 : al * a0;
    a1 = a1 >= 0.f ? a1 : al * a1;
    A[i]     = (_Float16)a0;
    A[i + 8] = (_Float16)a1;
  }
  v16h B0, B1;
#pragma unroll
  for (int i = 0; i < 8; ++i) {
    int k0 = kb + i;
    int k1 = kb + 16 + i;
    B0[i]     = w2h[k0 * HDIM + ncol];
    B0[i + 8] = w2h[k1 * HDIM + ncol];
    B1[i]     = w2h[k0 * HDIM + ncol + 16];
    B1[i + 8] = w2h[k1 * HDIM + ncol + 16];
  }
  v8f c = {};
  v8f t0 = __builtin_amdgcn_wmma_f32_16x16x32_f16(false, A, false, B0,
                                                  (short)0, c, false, false);
  v8f t1 = __builtin_amdgcn_wmma_f32_16x16x32_f16(false, A, false, B1,
                                                  (short)0, c, false, false);
#endif

  float w0a = w0[ncol], w0b = w0[ncol + 16];
  float va  = vvec[ncol], vb = vvec[ncol + 16];
  float p[8], q[8];
#pragma unroll
  for (int v = 0; v < 8; ++v) {
    p[v] = fmaf(t0[v], w0a, t1[v] * w0b);
    q[v] = fmaf(t0[v], va,  t1[v] * vb);
  }
#pragma unroll
  for (int mask = 1; mask <= 8; mask <<= 1) {
#pragma unroll
    for (int v = 0; v < 8; ++v) {
      p[v] += __shfl_xor(p[v], mask, 32);
      q[v] += __shfl_xor(q[v], mask, 32);
    }
  }
  if ((lane & 15) == 0) {
    int mb = half * 8;
#pragma unroll
    for (int v = 0; v < 8; ++v) {
      int r = row0 + mb + v;
      if (r < n) {
        aPart[r] = p[v] + cA;   // emb.w0 + b0
        sArr[r]  = q[v] + cS;   // emb.v
      }
    }
  }
}

// ---------------------------------------------------------------------------
// Kernel 6: one scalar atomic per edge: escore[dst] += norm_e * s[src]
// ---------------------------------------------------------------------------
__global__ void k_edge_scatter(const int* __restrict__ ei,
                               const float* __restrict__ ew,
                               const float* __restrict__ deg,
                               const float* __restrict__ sArr,
                               float* __restrict__ esc,
                               int e_total) {
  int e = blockIdx.x * blockDim.x + threadIdx.x;
  if (e >= e_total) return;
  int s = ei[e];
  int d = ei[e_total + e];
  float nrm = ew[e] * rsqrtf((deg[s] + 1.f) * (deg[d] + 1.f));
  atomicAdd(&esc[d], nrm * sArr[s]);
}

// ---------------------------------------------------------------------------
// Kernel 7: out = aPart + escore + s/deg + (bf + bg.wf)
// ---------------------------------------------------------------------------
__global__ void k_finalize(const float* __restrict__ aPart,
                           const float* __restrict__ esc,
                           const float* __restrict__ sArr,
                           const float* __restrict__ deg,
                           const float* __restrict__ stats,
                           float* __restrict__ out, int n) {
  int i = blockIdx.x * blockDim.x + threadIdx.x;
  if (i >= n) return;
  out[i] = aPart[i] + esc[i] + sArr[i] / (deg[i] + 1.f) + stats[162];
}

// ---------------------------------------------------------------------------
extern "C" void kernel_launch(void* const* d_in, const int* in_sizes, int n_in,
                              void* d_out, int out_size, void* d_ws, size_t ws_size,
                              hipStream_t stream) {
  const float* x     = (const float*)d_in[0];
  const int*   ei    = (const int*)d_in[1];
  const float* face  = (const float*)d_in[2];
  const float* W1    = (const float*)d_in[3];
  const float* b1    = (const float*)d_in[4];
  const float* gamma = (const float*)d_in[5];
  const float* beta  = (const float*)d_in[6];
  const float* alpha = (const float*)d_in[7];
  const float* W2    = (const float*)d_in[8];
  const float* b2    = (const float*)d_in[9];
  const float* w0    = (const float*)d_in[10];
  const float* b0    = (const float*)d_in[11];
  const float* Wg    = (const float*)d_in[12];
  const float* bg    = (const float*)d_in[13];
  const float* wf    = (const float*)d_in[14];
  const float* bf    = (const float*)d_in[15];

  int N = in_sizes[0] / 2;
  int E = in_sizes[1] / 2;
  int F = in_sizes[2] / N;

  float* ws = (float*)d_ws;
  size_t o_h   = 0;
  size_t o_ew  = o_h  + (size_t)N * HDIM;
  size_t o_inv = o_ew + (size_t)E;
  size_t o_deg = o_inv + (size_t)N;
  size_t o_esc = o_deg + (size_t)N;   // deg and esc adjacent -> one memset
  size_t o_s   = o_esc + (size_t)N;
  size_t o_a   = o_s   + (size_t)N;
  size_t o_st  = o_a   + (size_t)N;
  size_t o_w2h = o_st  + 192;         // 512 floats hold 1024 halves

  float*    h     = ws + o_h;
  float*    ew    = ws + o_ew;
  float*    invn  = ws + o_inv;
  float*    deg   = ws + o_deg;
  float*    esc   = ws + o_esc;
  float*    sArr  = ws + o_s;
  float*    aPart = ws + o_a;
  float*    stats = ws + o_st;
  _Float16* w2h   = (_Float16*)(ws + o_w2h);
  float*    out   = (float*)d_out;

  hipMemsetAsync(deg,   0, 2 * (size_t)N * sizeof(float), stream);  // deg + esc
  hipMemsetAsync(stats, 0, 64 * sizeof(float), stream);

  dim3 b256(256);
  k_h_stats      <<<dim3((N + 255) / 256), b256, 0, stream>>>(x, W1, b1, h, stats, N);
  k_face_inv_norm<<<dim3((N + 7) / 8),     b256, 0, stream>>>(face, invn, N, F / 4);
  k_setup        <<<dim3(1), dim3(32),     0, stream>>>(gamma, beta, W2, b2, w0, b0,
                                                        Wg, bg, wf, bf, stats, w2h,
                                                        1.0f / (float)N);
  k_edge_dot     <<<dim3((E + 7) / 8),     b256, 0, stream>>>(ei, face, invn, ew, deg, E, F / 4);
  k_emb_wmma     <<<dim3((N + 127) / 128), b256, 0, stream>>>(h, stats, W2, w2h, w0, alpha,
                                                              aPart, sArr, N);
  k_edge_scatter <<<dim3((E + 255) / 256), b256, 0, stream>>>(ei, ew, deg, sArr, esc, E);
  k_finalize     <<<dim3((N + 255) / 256), b256, 0, stream>>>(aPart, esc, sArr, deg, stats, out, N);

  (void)n_in; (void)out_size; (void)ws_size;
}
